// HMMModel_56109452755515
// MI455X (gfx1250) — matrix operand order repaired and checked by reading, not compiled, not run on hardware
//
#include <hip/hip_runtime.h>
#include <hip/hip_bf16.h>

// ---------------------------------------------------------------------------
// HMM posteriors on MI455X (gfx1250, wave32, WMMA).
//   emit = relu(obs @ w1^T + b1) @ w2^T + b2          -> f16 WMMA GEMMs
//   alpha/beta log-space scans: per step a 32x128 @ 128x128 GEMM in
//   probability space (exp(prev-max) @ softmax(trans)) done with WMMA,
//   transition matrix pinned in LDS as f16. Emission rows are streamed one
//   step ahead via async-to-LDS copies (ASYNCcnt) or prefetch fallback.
//   posterior = softmax_s(alpha+beta)   (logZ cancels)
// ---------------------------------------------------------------------------

constexpr int kB = 32;      // batch
constexpr int kT = 4096;    // time steps
constexpr int kD = 64;      // obs dim
constexpr int kH = 512;     // hidden (4*S)
constexpr int kS = 128;     // states

typedef _Float16 h8  __attribute__((ext_vector_type(8)));
typedef _Float16 h16 __attribute__((ext_vector_type(16)));
typedef float    f8  __attribute__((ext_vector_type(8)));
typedef int      v4i __attribute__((ext_vector_type(4)));

// -------- optional CDNA5 async-to-LDS path (guarded; fallback = prefetch) --
#if defined(__has_builtin)
#  if __has_builtin(__builtin_amdgcn_global_load_async_to_lds_b128)
#    define ASYNC_EMIT 1
#  endif
#endif
#ifndef ASYNC_EMIT
#  define ASYNC_EMIT 0
#endif

#if ASYNC_EMIT
typedef __attribute__((address_space(1))) v4i* g_v4i_ptr;  // global
typedef __attribute__((address_space(3))) v4i* l_v4i_ptr;  // LDS

__device__ __forceinline__ void async_copy16(const float* g, float* l) {
  __builtin_amdgcn_global_load_async_to_lds_b128(
      (g_v4i_ptr)g, (l_v4i_ptr)l, 0, 0);
}
__device__ __forceinline__ void wait_async_le2() {
#if __has_builtin(__builtin_amdgcn_s_wait_asynccnt)
  __builtin_amdgcn_s_wait_asynccnt(2);
#else
  asm volatile("s_wait_asynccnt 2" ::: "memory");
#endif
}
#endif

// A-matrix fragment (16xK tile, K-slice of 32) from row-major f16 [row][k].
// Lane L (row = L&15) holds contiguous K = 16*(L>=16) + e  -> one 32B chunk
// (same k order as load_b_frag).
__device__ __forceinline__ h16 load_a_frag(const _Float16* buf, int ld,
                                           int row_base, int kbase) {
  const int lane = threadIdx.x & 31;
  const _Float16* p = buf + (size_t)(row_base + (lane & 15)) * ld
                          + kbase + ((lane >> 4) << 4);
  return *(const h16*)(p);
}

// B-matrix fragment (Kx16 tile) from row-major f16 B^T i.e. [n][k].
// This code's k order (as load_a_frag): lane L holds column n=L&15, contiguous
// K = 16*(L>=16) + e  -> one 32B chunk.
__device__ __forceinline__ h16 load_b_frag(const _Float16* buf, int ld,
                                           int n_base, int kbase) {
  const int lane = threadIdx.x & 31;
  const _Float16* p = buf + (size_t)(n_base + (lane & 15)) * ld
                          + kbase + ((lane >> 4) << 4);
  return *(const h16*)(p);
}

__device__ __forceinline__ f8 wmma16(h16 a, h16 b, f8 c) {
  return __builtin_amdgcn_wmma_f32_16x16x32_f16(false, a, false, b,
                                                (short)0, c, false, false);
}

// ---------------------------------------------------------------------------
// Prep: f16 weights + row-softmax transition matrices, stored transposed
// (B^T layout) for the WMMA B operand.
//   TfT[s][i] = softmax_row_i(trans)[s]        (forward)
//   TbT[s][i] = softmax_row_i(trans^T)[s] = exp(trans[s][i] - colLSE_i)
// ---------------------------------------------------------------------------
__global__ __launch_bounds__(128)
void hmm_prep(const float* __restrict__ trans,
              const float* __restrict__ w1, const float* __restrict__ w2,
              _Float16* __restrict__ w1h, _Float16* __restrict__ w2h,
              _Float16* __restrict__ TfT, _Float16* __restrict__ TbT) {
  const int tid = threadIdx.x;           // 128 threads
  for (int i = tid; i < kH * kD; i += 128) w1h[i] = (_Float16)w1[i];
  for (int i = tid; i < kS * kH; i += 128) w2h[i] = (_Float16)w2[i];

  // forward: thread tid owns row i=tid of trans
  {
    const float* r = trans + (size_t)tid * kS;
    float mx = r[0];
    for (int s = 1; s < kS; ++s) mx = fmaxf(mx, r[s]);
    float sum = 0.f;
    for (int s = 0; s < kS; ++s) sum += __expf(r[s] - mx);
    const float inv = 1.0f / sum;
    for (int s = 0; s < kS; ++s)
      TfT[(size_t)s * kS + tid] = (_Float16)(__expf(r[s] - mx) * inv);
  }
  // backward: thread tid owns column i=tid of trans
  {
    float mx = trans[tid];
    for (int s = 1; s < kS; ++s) mx = fmaxf(mx, trans[(size_t)s * kS + tid]);
    float sum = 0.f;
    for (int s = 0; s < kS; ++s) sum += __expf(trans[(size_t)s * kS + tid] - mx);
    const float inv = 1.0f / sum;
    for (int s = 0; s < kS; ++s)
      TbT[(size_t)s * kS + tid] =
          (_Float16)(__expf(trans[(size_t)s * kS + tid] - mx) * inv);
  }
}

// ---------------------------------------------------------------------------
// Emission MLP: 16 (t,b)-rows per block, 8 waves.
//   stage 1: 16x512 hidden tile (32 WMMA n-tiles, K=64) -> relu -> LDS f16
//   stage 2: 16x128 emit tile   (8 WMMA n-tiles,  K=512) -> +b2 -> global f32
// emit layout [t][b][s], row r = t*32 + b.
// ---------------------------------------------------------------------------
__global__ __launch_bounds__(256)
void hmm_emit(const float* __restrict__ obs,
              const _Float16* __restrict__ w1h,
              const _Float16* __restrict__ w2h,
              const float* __restrict__ b1, const float* __restrict__ b2,
              float* __restrict__ emit) {
  __shared__ _Float16 sA[16 * kD];     // obs tile, f16
  __shared__ _Float16 sHid[16 * kH];   // hidden tile, f16 (16 KB)

  const int tid  = threadIdx.x;
  const int wave = tid >> 5;
  const int lane = tid & 31;
  const int r0   = blockIdx.x * 16;

  { // load 16x64 obs tile -> f16 LDS (obs is [b][t][d])
    const int i = tid >> 4, c = (tid & 15) * 4;
    const int r = r0 + i, t = r >> 5, b = r & 31;
    float4 v = *(const float4*)(obs + ((size_t)b * kT + t) * kD + c);
    _Float16* dst = &sA[i * kD + c];
    dst[0] = (_Float16)v.x; dst[1] = (_Float16)v.y;
    dst[2] = (_Float16)v.z; dst[3] = (_Float16)v.w;
  }
  __syncthreads();

  const h16 a0 = load_a_frag(sA, kD, 0, 0);
  const h16 a1 = load_a_frag(sA, kD, 0, 32);
  const int rbase = (lane >> 4) << 3;

#pragma unroll
  for (int q = 0; q < 4; ++q) {
    const int nt = wave + q * 8;                 // 0..31
    f8 acc = {};
    acc = wmma16(a0, load_b_frag(w1h, kD, nt * 16, 0), acc);
    acc = wmma16(a1, load_b_frag(w1h, kD, nt * 16, 32), acc);
    const int h = nt * 16 + (lane & 15);
    const float bias = b1[h];
#pragma unroll
    for (int r = 0; r < 8; ++r) {
      float v = fmaxf(acc[r] + bias, 0.0f);
      sHid[(rbase + r) * kH + h] = (_Float16)v;
    }
  }
  __syncthreads();

  f8 acc = {};
#pragma unroll
  for (int kb = 0; kb < kH; kb += 32)
    acc = wmma16(load_a_frag(sHid, kH, 0, kb),
                 load_b_frag(w2h, kH, wave * 16, kb), acc);
  const int s = wave * 16 + (lane & 15);
  const float bias = b2[s];
#pragma unroll
  for (int r = 0; r < 8; ++r)
    emit[(size_t)(r0 + rbase + r) * kS + s] = acc[r] + bias;
}

// ---------------------------------------------------------------------------
// Scan: grid = 4 blocks (dir = blockIdx>>1, batch half = blockIdx&1).
// Each block: persistent 8-wave loop over 4096 steps, 3 barriers/step.
//   m_b = max_i prev[b,i];  P = exp(prev-m) (f16)
//   sums = P @ T            (4 chained v_wmma, T pinned in LDS, B-frags hoisted)
//   nxt  = m_b + log(sums) + emit[t]  -> LDS prev + global log_alpha/log_beta
// Emission rows double-buffered one step ahead (async-to-LDS or prefetch).
// ---------------------------------------------------------------------------
__global__ __launch_bounds__(256)
void hmm_scan(const float* __restrict__ emit,
              const _Float16* __restrict__ TfT,
              const _Float16* __restrict__ TbT,
              float* __restrict__ log_alpha, float* __restrict__ log_beta) {
  __shared__ _Float16 sTT[kS * kS];    // 32 KB transition (B^T layout)
  __shared__ float    sPrev[16 * kS];  // 8 KB
  __shared__ _Float16 sP[16 * kS];     // 4 KB
  __shared__ float    sM[16];
  __shared__ float    sPart[16 * 16];
#if ASYNC_EMIT
  __shared__ float    sEmit[2][16 * kS];  // 16 KB double buffer
#endif

  const int dir  = blockIdx.x >> 1;
  const int bh   = blockIdx.x & 1;
  const int tid  = threadIdx.x;
  const int wave = tid >> 5;
  const int lane = tid & 31;

  const _Float16* TT = dir ? TbT : TfT;
  float* outbuf = dir ? log_beta : log_alpha;

  for (int i = tid; i < kS * kS; i += 256) sTT[i] = TT[i];
  for (int i = tid; i < 16 * kS; i += 256) sPrev[i] = 0.0f;   // h0 = zeros
  __syncthreads();

  // transition B-fragments are loop-invariant -> hoist into VGPRs
  h16 bf0 = load_b_frag(sTT, kS, wave * 16, 0);
  h16 bf1 = load_b_frag(sTT, kS, wave * 16, 32);
  h16 bf2 = load_b_frag(sTT, kS, wave * 16, 64);
  h16 bf3 = load_b_frag(sTT, kS, wave * 16, 96);

  const int row = tid >> 4;   // 0..15 (local batch row)
  const int grp = tid & 15;   // 8 elements each
  const int col = wave * 16 + (lane & 15);
  const int rb  = (lane >> 4) << 3;

  // emission row base (this block's 16-batch half) for time t
  auto emit_row = [&](int t) {
    return emit + ((size_t)t * kB + bh * 16) * kS;
  };

#if ASYNC_EMIT
  { // preload step 0 into buffer 0
    const float* g = emit_row(dir ? kT - 1 : 0);
    async_copy16(g + tid * 4, &sEmit[0][tid * 4]);
    async_copy16(g + (256 + tid) * 4, &sEmit[0][(256 + tid) * 4]);
  }
#endif

  for (int step = 0; step < kT; ++step) {
    const int t = dir ? (kT - 1 - step) : step;

    // (0) stream next emission row one step ahead
    if (step + 1 < kT) {
      const int tn = dir ? (t - 1) : (t + 1);
      const float* g = emit_row(tn);
#if ASYNC_EMIT
      float* lbuf = &sEmit[(step + 1) & 1][0];
      async_copy16(g + tid * 4, lbuf + tid * 4);
      async_copy16(g + (256 + tid) * 4, lbuf + (256 + tid) * 4);
#else
      __builtin_prefetch(g + tid * 8, 0, 0);
#endif
    }

    // (1) per-row partial max of prev (16 partials per row)
    const float* pr = &sPrev[row * kS + grp * 8];
    float m0 = pr[0];
#pragma unroll
    for (int j = 1; j < 8; ++j) m0 = fmaxf(m0, pr[j]);
    sPart[row * 16 + grp] = m0;
    __syncthreads();

    // (2) every thread finishes its row max locally (cheaper than a barrier),
    //     one thread per row publishes it for stage (4); P = exp(prev - m)
    float m = sPart[row * 16];
#pragma unroll
    for (int j = 1; j < 16; ++j) m = fmaxf(m, sPart[row * 16 + j]);
    if (grp == 0) sM[row] = m;
    _Float16* pp = &sP[row * kS + grp * 8];
#pragma unroll
    for (int j = 0; j < 8; ++j) pp[j] = (_Float16)__expf(pr[j] - m);
#if ASYNC_EMIT
    wait_async_le2();   // current step's emission row landed in LDS
#endif
    __syncthreads();

    // (3) sums = P @ T  (16x128 @ 128x16 tile per wave)
    f8 acc = {};
    acc = wmma16(load_a_frag(sP, kS, 0, 0),  bf0, acc);
    acc = wmma16(load_a_frag(sP, kS, 0, 32), bf1, acc);
    acc = wmma16(load_a_frag(sP, kS, 0, 64), bf2, acc);
    acc = wmma16(load_a_frag(sP, kS, 0, 96), bf3, acc);

    // (4) nxt = m + log(sums) + emit[t]; write prev + global
#if ASYNC_EMIT
    const float* erow = &sEmit[step & 1][0];
#else
    const float* erow = emit_row(t);
#endif
    float* orow = outbuf + (size_t)t * kB * kS;
#pragma unroll
    for (int r = 0; r < 8; ++r) {
      const int mr = rb + r;                       // local batch row
      const int gb = bh * 16 + mr;                 // global batch
      float v = sM[mr] + __logf(acc[r]) + erow[(size_t)mr * kS + col];
      sPrev[mr * kS + col] = v;
      orow[(size_t)gb * kS + col] = v;
    }
    __syncthreads();
  }
}

// ---------------------------------------------------------------------------
// Posterior: softmax_s(alpha + beta), one wave per (t,b) row,
// output transposed to [b][t][s].
// ---------------------------------------------------------------------------
__global__ __launch_bounds__(256)
void hmm_post(const float* __restrict__ la, const float* __restrict__ lb,
              float* __restrict__ out) {
  const int gw   = (int)((blockIdx.x * blockDim.x + threadIdx.x) >> 5);
  const int lane = threadIdx.x & 31;
  const size_t base = (size_t)gw * kS + lane * 4;

  float4 a = *(const float4*)(la + base);
  float4 b = *(const float4*)(lb + base);
  float g[4] = {a.x + b.x, a.y + b.y, a.z + b.z, a.w + b.w};

  float mx = fmaxf(fmaxf(g[0], g[1]), fmaxf(g[2], g[3]));
#pragma unroll
  for (int off = 16; off > 0; off >>= 1) mx = fmaxf(mx, __shfl_xor(mx, off, 32));

  float e[4], sum = 0.f;
#pragma unroll
  for (int j = 0; j < 4; ++j) { e[j] = __expf(g[j] - mx); sum += e[j]; }
#pragma unroll
  for (int off = 16; off > 0; off >>= 1) sum += __shfl_xor(sum, off, 32);
  const float inv = 1.0f / sum;

  const int t = gw >> 5, bi = gw & 31;
  float4 o = {e[0] * inv, e[1] * inv, e[2] * inv, e[3] * inv};
  *(float4*)(out + ((size_t)bi * kT + t) * kS + lane * 4) = o;
}

// ---------------------------------------------------------------------------
extern "C" void kernel_launch(void* const* d_in, const int* in_sizes, int n_in,
                              void* d_out, int out_size, void* d_ws,
                              size_t ws_size, hipStream_t stream) {
  const float* obs   = (const float*)d_in[0];   // (B,T,D)
  const float* trans = (const float*)d_in[1];   // (S,S)
  const float* w1    = (const float*)d_in[2];   // (4S,D)
  const float* b1    = (const float*)d_in[3];   // (4S)
  const float* w2    = (const float*)d_in[4];   // (S,4S)
  const float* b2    = (const float*)d_in[5];   // (S)
  // d_in[6] init_logits: unused by the reference computation.
  float* out = (float*)d_out;

  char* ws = (char*)d_ws;
  const size_t nTBS = (size_t)kT * kB * kS * sizeof(float);  // 64 MB
  float*    emitb = (float*)(ws);
  float*    alpha = (float*)(ws + nTBS);
  float*    beta  = (float*)(ws + 2 * nTBS);
  _Float16* w1h   = (_Float16*)(ws + 3 * nTBS);
  _Float16* w2h   = (_Float16*)((char*)w1h + (size_t)kH * kD * 2);
  _Float16* TfT   = (_Float16*)((char*)w2h + (size_t)kS * kH * 2);
  _Float16* TbT   = (_Float16*)((char*)TfT + (size_t)kS * kS * 2);

  hmm_prep<<<1, 128, 0, stream>>>(trans, w1, w2, w1h, w2h, TfT, TbT);
  hmm_emit<<<(kT * kB) / 16, 256, 0, stream>>>(obs, w1h, w2h, b1, b2, emitb);
  hmm_scan<<<4, 256, 0, stream>>>(emitb, TfT, TbT, alpha, beta);
  hmm_post<<<(kT * kB) / 8, 256, 0, stream>>>(alpha, beta, out);
}
